// ConceptCodebook_42477226557671
// MI455X (gfx1250) — matrix operand order repaired and checked by reading, not compiled, not run on hardware
//
#include <hip/hip_runtime.h>

// ---- CDNA5 (gfx1250) vector types for WMMA ----
typedef _Float16     v16h __attribute__((ext_vector_type(16)));
typedef float        v8f  __attribute__((ext_vector_type(8)));
typedef unsigned int v4u  __attribute__((ext_vector_type(4)));

#define D_DIM   768     // feature dim
#define M_TILE  128     // rows (z vectors) per workgroup (8 waves x 16 rows)
#define N_TILE  64      // codes per K-tile iteration
#define DEPTH   64      // depth per staging step = 2 WMMA k-chunks of 32
#define NSTEPS  (D_DIM / DEPTH)   // 12
#define SA      72      // LDS row stride in halves (144B, 16B aligned, padded)
#define SB      72

// A-fragment (16x32 f16, M x K): lane l -> row M=l&15; half h=l>>4.
// ISA layout: halves 0..7 = K[8h..8h+7], halves 8..15 = K[16+8h..16+8h+7].
__device__ inline v16h a_frag_lds(const _Float16* rowp, int h) {
    union { v4u u[2]; v16h f; } U;
    U.u[0] = *(const v4u*)(rowp + 8 * h);
    U.u[1] = *(const v4u*)(rowp + 16 + 8 * h);
    return U.f;
}

// B-fragment (32x16 f16, K x N): lane l -> col N=l&15; half h=l>>4.
// ISA layout: halves 0..7 = K[8h..8h+7], halves 8..15 = K[16+8h..16+8h+7].
__device__ inline v16h b_frag_lds(const _Float16* rowp, int h) {
    union { v4u u[2]; v16h f; } U;
    U.u[0] = *(const v4u*)(rowp + 8 * h);
    U.u[1] = *(const v4u*)(rowp + 16 + 8 * h);
    return U.f;
}

// Stage 32 fp32 -> fp16 into LDS (A path needs format conversion).
__device__ inline void stage_a(_Float16* dst, const float* src) {
    const float4* s = (const float4*)src;
#pragma unroll
    for (int i = 0; i < 8; ++i) {
        float4 v = s[i];
        dst[4 * i + 0] = (_Float16)v.x; dst[4 * i + 1] = (_Float16)v.y;
        dst[4 * i + 2] = (_Float16)v.z; dst[4 * i + 3] = (_Float16)v.w;
    }
}

// Stage 16 fp16 (32B) global -> LDS via CDNA5 async copy (ASYNCcnt-tracked).
__device__ inline void stage_b_async(_Float16* dst, const _Float16* src) {
    unsigned l0 = (unsigned)(unsigned long long)dst;   // LDS byte offset
    asm volatile("global_load_async_to_lds_b128 %0, %1, off"
                 :: "v"(l0), "v"((unsigned long long)src) : "memory");
    asm volatile("global_load_async_to_lds_b128 %0, %1, off"
                 :: "v"(l0 + 16u), "v"((unsigned long long)(src + 8)) : "memory");
}

__device__ inline void wait_async0() {
    asm volatile("s_wait_asynccnt 0x0" ::: "memory");
}

// ---- Kernel 0: pre-convert codebook to fp16 (reused by all 128 WGs) ----
__global__ __launch_bounds__(256)
void vq_emb_cvt(const float* __restrict__ emb, _Float16* __restrict__ embH) {
    long i = ((long)blockIdx.x * blockDim.x + threadIdx.x) * 4;
    float4 v = *(const float4*)(emb + i);
    embH[i + 0] = (_Float16)v.x; embH[i + 1] = (_Float16)v.y;
    embH[i + 2] = (_Float16)v.z; embH[i + 3] = (_Float16)v.w;
}

// ---- Kernel 1: per-code squared norms ||e_k||^2 (one wave32 per code) ----
__global__ __launch_bounds__(256) void vq_enorm(const float* __restrict__ emb,
                                                float* __restrict__ eN) {
    int k    = blockIdx.x * 8 + (threadIdx.x >> 5);
    int lane = threadIdx.x & 31;
    const float* p = emb + (long)k * D_DIM;
    float s = 0.f;
    for (int d = lane; d < D_DIM; d += 32) { float v = p[d]; s += v * v; }
#pragma unroll
    for (int off = 16; off >= 1; off >>= 1) s += __shfl_xor(s, off);
    if (lane == 0) eN[k] = s;
}

// ---- Kernel 2: fused distance GEMM (f16 WMMA, f32 accum) + running argmin ----
// Double-buffered LDS (one barrier/step), batched fragment loads (8 WMMAs
// back-to-back), async global->LDS for the fp16 codebook tiles, per-lane
// branchless running argmin reduced cross-lane once at the end.
__global__ __launch_bounds__(256)
void vq_dist_argmin(const float* __restrict__ z, const _Float16* __restrict__ embH,
                    const float* __restrict__ eNorm, int* __restrict__ idsInt,
                    float* __restrict__ idsF, int K) {
    __shared__ alignas(16) _Float16 As[2][M_TILE * SA];
    __shared__ alignas(16) _Float16 Bs[2][N_TILE * SB];

    const int tid  = threadIdx.x;
    const int lane = tid & 31;
    const int w    = tid >> 5;      // wave 0..7 -> rows w*16 .. w*16+15
    const int h    = lane >> 4;
    const int l15  = lane & 15;

    const long blockRow = (long)blockIdx.x * M_TILE;

    // Per-lane running minimum: slot j covers row (w*16 + j + 8*h); each lane
    // tracks the min over columns congruent to l15 (mod 16).
    float runV[8];
    int   runI[8];
#pragma unroll
    for (int j = 0; j < 8; ++j) { runV[j] = 3.4e38f; runI[j] = 0; }

    // Staging work assignment (constant per thread).
    const int ar    = tid >> 1;          // A row 0..127
    const int asegF = (tid & 1) * 32;    // 32-float depth segment
    const int bc    = tid >> 2;          // B row (code) 0..63
    const int bsegH = (tid & 3) * 16;    // 16-half depth segment

    const float* zRow = z + (blockRow + ar) * D_DIM + asegF;
    const unsigned aOff = ar * SA + asegF;
    const unsigned bOff = bc * SB + bsegH;

    const int numCT = K / N_TILE;
    for (int ct = 0; ct < numCT; ++ct) {
        // Prefetch next fp16 codebook tile (global_prefetch_b8).
        if (ct + 1 < numCT)
            __builtin_prefetch(embH + ((long)(ct + 1) * N_TILE + (tid & (N_TILE - 1))) * D_DIM, 0, 1);

        const _Float16* eRowH = embH + ((long)ct * N_TILE + bc) * D_DIM + bsegH;
        v8f acc[4] = {};   // fp32 accumulators: 4 x (16x16) tiles, all 64 cols

        // Prologue: stage step 0 into buffer 0.
        stage_a(As[0] + aOff, zRow);
        stage_b_async(Bs[0] + bOff, eRowH);
        wait_async0();
        __syncthreads();

        for (int s = 0; s < NSTEPS; ++s) {
            const int cur = s & 1;
            if (s + 1 < NSTEPS) {   // stage next step into alternate buffer
                stage_a(As[cur ^ 1] + aOff, zRow + (s + 1) * DEPTH);
                stage_b_async(Bs[cur ^ 1] + bOff, eRowH + (s + 1) * DEPTH);
            }

            // Batched fragment loads, then 8 WMMAs back-to-back.
            const _Float16* aRow = As[cur] + (w * 16 + l15) * SA;
            v16h afr[2];
            v16h bfr[2][4];
            afr[0] = a_frag_lds(aRow, h);
            afr[1] = a_frag_lds(aRow + 32, h);
#pragma unroll
            for (int kk = 0; kk < 2; ++kk)
#pragma unroll
                for (int ni = 0; ni < 4; ++ni)
                    bfr[kk][ni] = b_frag_lds(Bs[cur] + (ni * 16 + l15) * SB + kk * 32, h);
#pragma unroll
            for (int kk = 0; kk < 2; ++kk)
#pragma unroll
                for (int ni = 0; ni < 4; ++ni)
                    acc[ni] = __builtin_amdgcn_wmma_f32_16x16x32_f16(
                        false, afr[kk], false, bfr[kk][ni], (short)0, acc[ni], false, false);

            if (s + 1 < NSTEPS) wait_async0();
            __syncthreads();
        }

        // Branchless per-lane argmin update: dist = ||e||^2 - 2*dot
        // (||z||^2 constant per row -> dropped). Candidate indices arrive in
        // strictly increasing order, so strict '<' == first occurrence.
        const int colB = ct * N_TILE + l15;
#pragma unroll
        for (int ni = 0; ni < 4; ++ni) {
            const int col = colB + ni * 16;
            const float en = eNorm[col];
#pragma unroll
            for (int j = 0; j < 8; ++j) {
                float d = en - 2.0f * acc[ni][j];
                bool t = d < runV[j];
                runV[j] = t ? d : runV[j];
                runI[j] = t ? col : runI[j];
            }
        }
    }

    // One-time cross-lane reduction within each 16-lane half (branchless).
#pragma unroll
    for (int j = 0; j < 8; ++j) {
        float v = runV[j]; int ix = runI[j];
#pragma unroll
        for (int off = 1; off < 16; off <<= 1) {
            float vo = __shfl_xor(v, off);
            int   io = __shfl_xor(ix, off);
            bool t = (vo < v) | ((vo == v) & (io < ix));   // tie -> smaller index
            v  = t ? vo : v;
            ix = t ? io : ix;
        }
        runV[j] = v; runI[j] = ix;
    }

    if (l15 == 0) {
#pragma unroll
        for (int j = 0; j < 8; ++j) {
            long row = blockRow + w * 16 + j + 8 * h;
            idsInt[row] = runI[j];
            idsF[row]   = (float)runI[j];
        }
    }
}

// ---- Kernel 3: z_q gather (exact fp32) + per-block loss partial sums ----
__global__ __launch_bounds__(256)
void vq_gather_loss(const float* __restrict__ z, const float* __restrict__ emb,
                    const int* __restrict__ ids, float* __restrict__ outQ,
                    float* __restrict__ partials, long total) {
    __shared__ float red[256];
    float acc = 0.f;
    const long stride = (long)gridDim.x * blockDim.x;
    for (long i = (long)blockIdx.x * blockDim.x + threadIdx.x; i < total; i += stride) {
        long n = i / D_DIM;
        int  d = (int)(i - n * D_DIM);
        int  id = ids[n];
        float q = emb[(long)id * D_DIM + d];
        outQ[i] = q;                    // z_q_st == z_q numerically
        float diff = z[i] - q;
        acc += diff * diff;
    }
    red[threadIdx.x] = acc;
    __syncthreads();
    for (int s = 128; s > 0; s >>= 1) {
        if (threadIdx.x < s) red[threadIdx.x] += red[threadIdx.x + s];
        __syncthreads();
    }
    if (threadIdx.x == 0) partials[blockIdx.x] = red[0];
}

// ---- Kernel 4: deterministic final loss reduction ----
__global__ void vq_finalize(const float* __restrict__ partials, int nb,
                            float* __restrict__ lossOut, float scale) {
    if (threadIdx.x == 0) {
        float s = 0.f;
        for (int i = 0; i < nb; ++i) s += partials[i];
        *lossOut = scale * s;
    }
}

extern "C" void kernel_launch(void* const* d_in, const int* in_sizes, int n_in,
                              void* d_out, int out_size, void* d_ws, size_t ws_size,
                              hipStream_t stream) {
    const float* z   = (const float*)d_in[0];   // [8,2048,768] fp32
    const float* emb = (const float*)d_in[1];   // [2048,768]  fp32
    const long N = in_sizes[0] / D_DIM;         // 16384 rows
    const int  K = in_sizes[1] / D_DIM;         // 2048 codes

    // d_out layout (flat, return order): z_q_st [N*D] | concept_ids [N] | vq_loss [1]
    float* outQ    = (float*)d_out;
    float* outIds  = outQ + N * (long)D_DIM;
    float* outLoss = outIds + N;

    // workspace: eNorm [K] f32 | idsInt [N] i32 | partials [256] f32 | embH [K*D] f16
    float*    eN       = (float*)d_ws;
    int*      idsInt   = (int*)(eN + K);
    float*    partials = (float*)(idsInt + N);
    _Float16* embH     = (_Float16*)(partials + 256);   // 16B-aligned offset

    const long kd = (long)K * D_DIM;
    vq_emb_cvt<<<(int)(kd / (256 * 4)), 256, 0, stream>>>(emb, embH);
    vq_enorm<<<K / 8, 256, 0, stream>>>(emb, eN);
    vq_dist_argmin<<<(int)(N / M_TILE), 256, 0, stream>>>(z, embH, eN, idsInt, outIds, K);

    const int  NB    = 192;
    const long total = N * (long)D_DIM;
    vq_gather_loss<<<NB, 256, 0, stream>>>(z, emb, idsInt, outQ, partials, total);
    vq_finalize<<<1, 32, 0, stream>>>(partials, NB, outLoss, 0.25f / (float)total);
}
